// TiltedRecurrentLayer_56212531970293
// MI455X (gfx1250) — matrix operand order repaired and checked by reading, not compiled, not run on hardware
//
#include <hip/hip_runtime.h>

#define UNITS 1024
#define FT    512
#define TIME  256
#define BATCH 128
#define OUT_T   (TIME + 1)        // 257
#define OUT_ROW (OUT_T * UNITS)   // 263168 floats per batch row
#define NBLOCKS 64                // persistent scan blocks (8 mtiles x 8 ngroups)

typedef __attribute__((ext_vector_type(16))) __bf16        v16bf;
typedef __attribute__((ext_vector_type(8)))  float         v8f;
typedef __attribute__((ext_vector_type(4)))  unsigned int  v4u;
typedef __attribute__((ext_vector_type(4)))  int           v4i;

#define AS1 __attribute__((address_space(1)))
#define AS3 __attribute__((address_space(3)))

#if defined(__has_builtin)
#if __has_builtin(__builtin_amdgcn_global_load_async_to_lds_b128)
#define HAVE_ASYNC_LDS 1
#endif
#endif
#ifndef HAVE_ASYNC_LDS
#define HAVE_ASYNC_LDS 0
#endif

union AFrag { v16bf v; v4u q[2]; };
union BFrag { v16bf v; v4u q[2]; };

__device__ __forceinline__ unsigned short f2bf(float f) {
    unsigned int x = __float_as_uint(f);
    x += 0x7FFFu + ((x >> 16) & 1u);   // round-to-nearest-even
    return (unsigned short)(x >> 16);
}

// ---------------------------------------------------------------------------
// Pack a KxU fp32 matrix (row-major, U==1024) into bf16 WMMA B-fragments.
// Fragment (n,kt) holds B[k = kt*32 + Klocal][u = n*16 + (lane&15)] with
// Klocal = 8*(lane>=16) + 2v + 8*(v>=4) (+1 in hi half of dword v).
// Storage layout: [n][kt][p=v>>2][lane][4 dwords]  -> every b128 LDS read of
// a fragment covers all 64 banks (2-cycle optimal), and global reads of one
// (kt,p) block are 512B contiguous per wave.
// ---------------------------------------------------------------------------
__global__ __launch_bounds__(256) void frag_convert(const float* __restrict__ src,
                                                    unsigned int* __restrict__ dst,
                                                    int nKT) {
    int idx  = blockIdx.x * 256 + threadIdx.x;
    int v    = idx & 7;
    int lane = (idx >> 3) & 31;
    int ktn  = idx >> 8;                       // n*nKT + kt
    int u     = (ktn / nKT) * 16 + (lane & 15);
    int kt    = ktn % nKT;
    int kbase = kt * 32 + 8 * (lane >> 4) + 2 * v + 8 * (v >> 2);
    unsigned int lo = f2bf(src[(size_t)kbase * UNITS + u]);
    unsigned int hi = f2bf(src[(size_t)(kbase + 1) * UNITS + u]);
    size_t dsti = ((((size_t)ktn * 2 + (v >> 2)) * 32 + lane) * 4) + (v & 3);
    dst[dsti] = (hi << 16) | lo;
}

__global__ __launch_bounds__(256) void zero_u32(unsigned int* __restrict__ p, int n) {
    int i = blockIdx.x * 256 + threadIdx.x;
    if (i < n) p[i] = 0u;
}

// state mirror init: sbuf[0][b][u] = bf16(x0[u]) for all b (65536 dwords)
__global__ __launch_bounds__(256) void init_state(const float* __restrict__ x0,
                                                  unsigned int* __restrict__ sb) {
    int i  = blockIdx.x * 256 + threadIdx.x;
    int u2 = i & 511;                          // dword within a 1024-elem row
    unsigned int lo = f2bf(x0[u2 * 2]);
    unsigned int hi = f2bf(x0[u2 * 2 + 1]);
    sb[i] = (hi << 16) | lo;
}

// ---------------------------------------------------------------------------
// h_in projection: out[b,t,u] = sum_f X[b,t,f]*R[f,u] + bias[u]
// ---------------------------------------------------------------------------
#define PROJ_STRIDE 520   // ushorts/row: 260 dwords; 260%64==4 (bank-tiled b128)

__global__ __launch_bounds__(256) void proj_kernel(const float* __restrict__ X,
                                                   const unsigned int* __restrict__ Rfrag,
                                                   const float* __restrict__ bias,
                                                   float* __restrict__ out) {
    __shared__ __align__(16) unsigned short As[16 * PROJ_STRIDE];
    const int mtile = blockIdx.x;
    const int tid   = threadIdx.x;

    // stage 16 x 512 fp32 -> bf16 LDS (2048 float4 chunks)
    const float4* src = (const float4*)(X + (size_t)mtile * 16 * FT);
    for (int c = tid; c < 2048; c += 256) {
        float4 f  = src[c];
        int row   = c >> 7;       // 128 float4 per row
        int col4  = c & 127;
        unsigned int* pp = (unsigned int*)&As[row * PROJ_STRIDE + col4 * 4];
        pp[0] = ((unsigned int)f2bf(f.y) << 16) | f2bf(f.x);
        pp[1] = ((unsigned int)f2bf(f.w) << 16) | f2bf(f.z);
    }
    __syncthreads();

    const int lane = tid & 31;
    const int wave = tid >> 5;
    const int half = lane >> 4;
    const int mn   = lane & 15;

    for (int i = 0; i < 8; ++i) {
        const int ntile = 8 * i + wave;
        v8f c = {};
        for (int kt = 0; kt < 16; ++kt) {
            AFrag a; BFrag b;
            const int kb = kt * 32 + 8 * half;
            a.q[0] = *(const v4u*)&As[mn * PROJ_STRIDE + kb];
            a.q[1] = *(const v4u*)&As[mn * PROJ_STRIDE + kb + 16];
            const v4u* bq = (const v4u*)Rfrag;
            size_t fb = ((size_t)(ntile * 16 + kt) * 2) * 32 + lane;
            b.q[0] = bq[fb];
            b.q[1] = bq[fb + 32];
            c = __builtin_amdgcn_wmma_f32_16x16x32_bf16(false, a.v, false, b.v,
                                                        (short)0, c, false, false);
        }
        const int u  = ntile * 16 + mn;
        const float bv = bias[u];
#pragma unroll
        for (int r = 0; r < 8; ++r) {
            int m  = mtile * 16 + r + 8 * half;   // flattened (b,t)
            int bb = m >> 8;
            int t  = m & 255;
            out[(size_t)bb * OUT_ROW + (size_t)t * UNITS + u] = c[r] + bv;
        }
    }
}

// ---------------------------------------------------------------------------
// Persistent recurrence scan: one launch, 256 steps, device-wide rel/acq
// barrier per step. W fragments LDS-resident (256 KB); bf16 state mirror is
// double-buffered in global and async-copied into LDS each step; exp(s)
// accumulates in registers; terminal row written at the end.
// ---------------------------------------------------------------------------
#define STEP_STRIDE 1032  // ushorts/row: 516 dwords; 516%64==4 (bank-tiled b128)

__global__ __launch_bounds__(256) void scan_kernel(float* __restrict__ out,
                                                   const unsigned int* __restrict__ Wfrag,
                                                   unsigned short* __restrict__ sb0,
                                                   unsigned short* __restrict__ sb1,
                                                   unsigned int* __restrict__ bar) {
    __shared__ __align__(16) unsigned short As[16 * STEP_STRIDE];   //  33024 B
    __shared__ __align__(16) unsigned int   Bs[8 * 32 * 32 * 8];    // 262144 B

    const int mtile = blockIdx.x & 7;
    const int ng    = blockIdx.x >> 3;
    const int tid   = threadIdx.x;
    const int lane  = tid & 31;
    const int wave  = tid >> 5;
    const int half  = lane >> 4;
    const int mn    = lane & 15;
    const int ntile = ng * 8 + wave;

    // one-time: copy this block's 8 B-fragment columns (64K dwords) into LDS
    {
        const v4u* src4 = (const v4u*)(Wfrag + (size_t)ng * 65536);
        v4u* dst4 = (v4u*)Bs;
        for (int i = tid; i < 16384; i += 256) dst4[i] = src4[i];
    }

    v8f eacc = {};   // running sum of exp(s) over t, per owned (b,u)

    for (int t = 0; t < TIME; ++t) {
        const unsigned short* sprev = (t & 1) ? sb1 : sb0;
        unsigned short*       snext = (t & 1) ? sb0 : sb1;

        // stage 16 x 1024 bf16 state rows -> LDS (2048 x 16B chunks)
        for (int c = tid; c < 2048; c += 256) {
            int row = c >> 7;          // 128 chunks per row
            int col = c & 127;
            const unsigned short* gsrc =
                sprev + (((size_t)(mtile * 16 + row)) << 10) + col * 8;
            unsigned short* ldst = &As[row * STEP_STRIDE + col * 8];
#if HAVE_ASYNC_LDS
            __builtin_amdgcn_global_load_async_to_lds_b128(
                (AS1 v4i*)gsrc, (AS3 v4i*)ldst, 0, 0);
#else
            *(v4u*)ldst = *(const v4u*)gsrc;
#endif
        }
#if HAVE_ASYNC_LDS
#if __has_builtin(__builtin_amdgcn_s_wait_asynccnt)
        __builtin_amdgcn_s_wait_asynccnt(0);
#else
        asm volatile("s_wait_asynccnt 0x0" ::: "memory");
#endif
#endif
        __syncthreads();

        v8f c = {};
        for (int kt = 0; kt < 32; ++kt) {
            AFrag a; BFrag b;
            const int kb = kt * 32 + 8 * half;
            a.q[0] = *(const v4u*)&As[mn * STEP_STRIDE + kb];
            a.q[1] = *(const v4u*)&As[mn * STEP_STRIDE + kb + 16];
            const v4u* bq = (const v4u*)Bs;
            int fb = ((wave * 32 + kt) * 2) * 32 + lane;
            b.q[0] = bq[fb];
            b.q[1] = bq[fb + 32];
            c = __builtin_amdgcn_wmma_f32_16x16x32_bf16(false, a.v, false, b.v,
                                                        (short)0, c, false, false);
        }

        const int u = ntile * 16 + mn;
        float* hs = out + (size_t)t * UNITS;
#pragma unroll
        for (int r = 0; r < 8; ++r) {
            int bb = mtile * 16 + r + 8 * half;
            size_t off = (size_t)bb * OUT_ROW + u;
            float s = tanhf(hs[off] + c[r]);
            hs[off] = s;                                 // fp32 output row t
            snext[((size_t)bb << 10) + u] = f2bf(s);     // bf16 state mirror
            eacc[r] += expf(s);                          // BETA == 1
        }

        // device-wide barrier: publish state t, wait for all 64 blocks
        __syncthreads();
        if (tid == 0) {
            __hip_atomic_fetch_add(&bar[t], 1u, __ATOMIC_RELEASE,
                                   __HIP_MEMORY_SCOPE_AGENT);
            while (__hip_atomic_load(&bar[t], __ATOMIC_ACQUIRE,
                                     __HIP_MEMORY_SCOPE_AGENT) < NBLOCKS) {
                __builtin_amdgcn_s_sleep(1);
            }
        }
        __syncthreads();
    }

    // terminal row: log(tanh(mean_t exp(s)))   (beta == 1)
    {
        const int u = ntile * 16 + mn;
#pragma unroll
        for (int r = 0; r < 8; ++r) {
            int bb = mtile * 16 + r + 8 * half;
            out[(size_t)bb * OUT_ROW + (size_t)TIME * UNITS + u] =
                logf(tanhf(eacc[r] * (1.0f / (float)TIME)));
        }
    }
}

extern "C" void kernel_launch(void* const* d_in, const int* in_sizes, int n_in,
                              void* d_out, int out_size, void* d_ws, size_t ws_size,
                              hipStream_t stream) {
    (void)in_sizes; (void)n_in; (void)out_size; (void)ws_size;
    const float* inputs = (const float*)d_in[0];   // [128,256,512]
    const float* R      = (const float*)d_in[1];   // [512,1024]
    const float* W      = (const float*)d_in[2];   // [1024,1024]
    const float* bias   = (const float*)d_in[3];   // [1024]
    const float* x0     = (const float*)d_in[4];   // [1024] (zeros)
    float* out = (float*)d_out;                    // [128,257,1024]

    // workspace: Wfrag 2MB | Rfrag 1MB | bar 1KB | sbuf0 256KB | sbuf1 256KB
    unsigned int* Wfrag = (unsigned int*)d_ws;             // 524288 dwords
    unsigned int* Rfrag = Wfrag + 524288;                  // 262144 dwords
    unsigned int* bar   = Rfrag + 262144;                  //    256 dwords
    unsigned int* sb0d  = bar + 256;                       //  65536 dwords
    unsigned int* sb1d  = sb0d + 65536;                    //  65536 dwords

    frag_convert<<<2048, 256, 0, stream>>>(W, Wfrag, 32);
    frag_convert<<<1024, 256, 0, stream>>>(R, Rfrag, 16);
    zero_u32<<<1, 256, 0, stream>>>(bar, TIME);
    init_state<<<256, 256, 0, stream>>>(x0, sb0d);

    // h_in = X @ R + bias, written into d_out[:, 0:256, :]
    proj_kernel<<<2048, 256, 0, stream>>>(inputs, Rfrag, bias, out);

    // single persistent kernel runs all 256 recurrence steps
    scan_kernel<<<NBLOCKS, 256, 0, stream>>>(out, Wfrag,
                                             (unsigned short*)sb0d,
                                             (unsigned short*)sb1d, bar);
}
